// MoEBertSelfOutput_38534446579846
// MI455X (gfx1250) — compile-verified, matrix operand order checked by reading, not run on hardware
//
#include <hip/hip_runtime.h>

// Problem constants (from reference): E=8, B=16, S=512, H=1024, fp32 in/out.
#define HD 1024
#define SQ 512
#define BQ 16
#define BM 128
#define BN 128
#define BK 32
#define KS 40  // padded LDS K-stride (elements) -> 80B rows, conflict-free b128 reads

typedef __attribute__((ext_vector_type(16))) __bf16 v16bf;
typedef __attribute__((ext_vector_type(8)))  float  v8f;

union V16 { v16bf v; uint4 q[2]; };

__device__ __forceinline__ unsigned short bf16_rne(float f) {
    unsigned int u = __float_as_uint(f);
    u += 0x7FFFu + ((u >> 16) & 1u);            // round-to-nearest-even
    return (unsigned short)(u >> 16);
}
__device__ __forceinline__ float bf16_to_f32(unsigned short h) {
    return __uint_as_float(((unsigned int)h) << 16);
}

__global__ __launch_bounds__(256)
void moe_gemm_bias_resid_kernel(const float* __restrict__ hidden,
                                const float* __restrict__ resid,
                                const int*   __restrict__ eidx,
                                const float* __restrict__ Wall,
                                const float* __restrict__ ball,
                                float*       __restrict__ out) {
    __shared__ unsigned short Ahi[BM * KS];
    __shared__ unsigned short Alo[BM * KS];
    __shared__ unsigned short Bhi[BN * KS];   // stored transposed: [n][k]
    __shared__ unsigned short Blo[BN * KS];

    const int tid = threadIdx.x;
    const int bz  = blockIdx.z;               // batch
    const int m0  = blockIdx.y * BM;          // seq tile
    const int n0  = blockIdx.x * BN;          // hidden-out tile
    const int e   = eidx[bz];

    const float* __restrict__ A  = hidden + (size_t)bz * SQ * HD;
    const float* __restrict__ Wp = Wall + (size_t)e * HD * HD;
    const float* __restrict__ bp = ball + (size_t)e * HD;

    const int lane  = tid & 31;
    const int wave  = tid >> 5;
    const int wmOff = (wave & 3) * 32;        // 4 waves along M
    const int wnOff = (wave >> 2) * 64;       // 2 waves along N
    const int rA    = lane & 15;
    const int kh    = (lane >> 4) << 3;       // 0 or 8: K-chunk base per lane half

    v8f acc[2][4];
#pragma unroll
    for (int ms = 0; ms < 2; ++ms)
#pragma unroll
        for (int ns = 0; ns < 4; ++ns)
            acc[ms][ns] = (v8f)0.0f;

    // Global->LDS staging assignments
    const int ac = (tid & 7) * 4;             // A: k offset (float4)
    const int ar = tid >> 3;                  // A: row base (0..31)
    const int bn = (tid & 31) * 4;            // B: n offset (float4)
    const int bk = tid >> 5;                  // B: k row base (0..7)

    for (int kk = 0; kk < HD; kk += BK) {
        if (kk + BK < HD) {
            __builtin_prefetch(A  + (size_t)(m0 + ar) * HD + (kk + BK) + ac, 0, 1);
            __builtin_prefetch(Wp + (size_t)(kk + BK + bk) * HD + n0 + bn, 0, 1);
        }
        __syncthreads();   // previous iteration's LDS reads complete

        // ---- A tile (BM x BK fp32) -> Ahi/Alo bf16, row-major [m][k]
#pragma unroll
        for (int it = 0; it < 4; ++it) {
            int r = ar + it * 32;
            float4 a4 = *reinterpret_cast<const float4*>(
                A + (size_t)(m0 + r) * HD + kk + ac);
            unsigned short h0 = bf16_rne(a4.x), h1 = bf16_rne(a4.y);
            unsigned short h2 = bf16_rne(a4.z), h3 = bf16_rne(a4.w);
            unsigned short l0 = bf16_rne(a4.x - bf16_to_f32(h0));
            unsigned short l1 = bf16_rne(a4.y - bf16_to_f32(h1));
            unsigned short l2 = bf16_rne(a4.z - bf16_to_f32(h2));
            unsigned short l3 = bf16_rne(a4.w - bf16_to_f32(h3));
            uint2 ph, pl;
            ph.x = (unsigned)h0 | ((unsigned)h1 << 16);
            ph.y = (unsigned)h2 | ((unsigned)h3 << 16);
            pl.x = (unsigned)l0 | ((unsigned)l1 << 16);
            pl.y = (unsigned)l2 | ((unsigned)l3 << 16);
            *reinterpret_cast<uint2*>(&Ahi[r * KS + ac]) = ph;
            *reinterpret_cast<uint2*>(&Alo[r * KS + ac]) = pl;
        }

        // ---- B tile (BK x BN fp32, row-major in W) -> transposed bf16 [n][k]
#pragma unroll
        for (int it = 0; it < 4; ++it) {
            int kr = bk + it * 8;
            float4 w4 = *reinterpret_cast<const float4*>(
                Wp + (size_t)(kk + kr) * HD + n0 + bn);
            float wv[4] = {w4.x, w4.y, w4.z, w4.w};
#pragma unroll
            for (int j = 0; j < 4; ++j) {
                unsigned short h = bf16_rne(wv[j]);
                unsigned short l = bf16_rne(wv[j] - bf16_to_f32(h));
                Bhi[(bn + j) * KS + kr] = h;
                Blo[(bn + j) * KS + kr] = l;
            }
        }
        __syncthreads();

        // ---- Fragments from LDS (16B ds_load_b128 per half-fragment)
        V16 fah[2], fal[2], fbh[4], fbl[4];
#pragma unroll
        for (int ms = 0; ms < 2; ++ms) {
            int row = wmOff + ms * 16 + rA;
            fah[ms].q[0] = *reinterpret_cast<const uint4*>(&Ahi[row * KS + kh]);
            fah[ms].q[1] = *reinterpret_cast<const uint4*>(&Ahi[row * KS + 16 + kh]);
            fal[ms].q[0] = *reinterpret_cast<const uint4*>(&Alo[row * KS + kh]);
            fal[ms].q[1] = *reinterpret_cast<const uint4*>(&Alo[row * KS + 16 + kh]);
        }
#pragma unroll
        for (int ns = 0; ns < 4; ++ns) {
            int col = wnOff + ns * 16 + rA;
            fbh[ns].q[0] = *reinterpret_cast<const uint4*>(&Bhi[col * KS + kh]);
            fbh[ns].q[1] = *reinterpret_cast<const uint4*>(&Bhi[col * KS + 16 + kh]);
            fbl[ns].q[0] = *reinterpret_cast<const uint4*>(&Blo[col * KS + kh]);
            fbl[ns].q[1] = *reinterpret_cast<const uint4*>(&Blo[col * KS + 16 + kh]);
        }

        // ---- bf16x3 split-precision WMMA: Ahi*Bhi + Ahi*Blo + Alo*Bhi
#pragma unroll
        for (int ms = 0; ms < 2; ++ms)
#pragma unroll
            for (int ns = 0; ns < 4; ++ns) {
                acc[ms][ns] = __builtin_amdgcn_wmma_f32_16x16x32_bf16(
                    false, fah[ms].v, false, fbh[ns].v, (short)0, acc[ms][ns], false, false);
                acc[ms][ns] = __builtin_amdgcn_wmma_f32_16x16x32_bf16(
                    false, fah[ms].v, false, fbl[ns].v, (short)0, acc[ms][ns], false, false);
                acc[ms][ns] = __builtin_amdgcn_wmma_f32_16x16x32_bf16(
                    false, fal[ms].v, false, fbh[ns].v, (short)0, acc[ms][ns], false, false);
            }
    }

    // ---- Epilogue: + bias[e] + residual, write x (pre-LayerNorm) to d_out
#pragma unroll
    for (int ns = 0; ns < 4; ++ns) {
        int n = n0 + wnOff + ns * 16 + rA;          // C/D: N = lane & 15
        float bv = bp[n];
#pragma unroll
        for (int ms = 0; ms < 2; ++ms) {
            int rbase = m0 + wmOff + ms * 16 + ((lane >> 4) << 3);  // M = vgpr + 8*(lane>=16)
#pragma unroll
            for (int j = 0; j < 8; ++j) {
                int srow = rbase + j;
                size_t off = ((size_t)bz * SQ + srow) * HD + n;
                out[off] = acc[ms][ns][j] + bv + resid[off];
            }
        }
    }
}

__device__ __forceinline__ float block_sum(float v, float* sbuf, int tid) {
#pragma unroll
    for (int o = 16; o > 0; o >>= 1) v += __shfl_xor(v, o, 32);
    if ((tid & 31) == 0) sbuf[tid >> 5] = v;
    __syncthreads();
    if (tid < 8) {
        float t = sbuf[tid];
        t += __shfl_xor(t, 4, 32);
        t += __shfl_xor(t, 2, 32);
        t += __shfl_xor(t, 1, 32);
        if (tid == 0) sbuf[8] = t;
    }
    __syncthreads();
    float r = sbuf[8];
    __syncthreads();
    return r;
}

__global__ __launch_bounds__(256)
void layernorm_inplace_kernel(float* __restrict__ x,
                              const float* __restrict__ gamma,
                              const float* __restrict__ beta) {
    __shared__ float sbuf[9];
    const int tid = threadIdx.x;
    const size_t row = blockIdx.x;
    float* __restrict__ xr = x + row * HD;

    float4 v = *reinterpret_cast<const float4*>(xr + tid * 4);
    float s = v.x + v.y + v.z + v.w;
    s = block_sum(s, sbuf, tid);
    const float mu = s * (1.0f / HD);

    float dx = v.x - mu, dy = v.y - mu, dz = v.z - mu, dw = v.w - mu;
    float ss = dx * dx + dy * dy + dz * dz + dw * dw;
    ss = block_sum(ss, sbuf, tid);
    const float inv = rsqrtf(ss * (1.0f / HD) + 1e-12f);

    float4 g = *reinterpret_cast<const float4*>(gamma + tid * 4);
    float4 bt = *reinterpret_cast<const float4*>(beta + tid * 4);
    float4 y;
    y.x = dx * inv * g.x + bt.x;
    y.y = dy * inv * g.y + bt.y;
    y.z = dz * inv * g.z + bt.z;
    y.w = dw * inv * g.w + bt.w;
    *reinterpret_cast<float4*>(xr + tid * 4) = y;
}

extern "C" void kernel_launch(void* const* d_in, const int* in_sizes, int n_in,
                              void* d_out, int out_size, void* d_ws, size_t ws_size,
                              hipStream_t stream) {
    (void)in_sizes; (void)n_in; (void)out_size; (void)d_ws; (void)ws_size;
    const float* hidden = (const float*)d_in[0];   // [B,S,H] f32
    const float* resid  = (const float*)d_in[1];   // [B,S,H] f32
    const int*   eidx   = (const int*)d_in[2];     // [B] i32
    const float* Wall   = (const float*)d_in[3];   // [E,H,H] f32
    const float* ball   = (const float*)d_in[4];   // [E,H] f32
    const float* gamma  = (const float*)d_in[5];   // [H] f32
    const float* beta   = (const float*)d_in[6];   // [H] f32
    float* out = (float*)d_out;                    // [B,S,H] f32

    dim3 grid(HD / BN, SQ / BM, BQ);               // 8 x 4 x 16
    moe_gemm_bias_resid_kernel<<<grid, 256, 0, stream>>>(hidden, resid, eidx,
                                                         Wall, ball, out);
    layernorm_inplace_kernel<<<dim3(BQ * SQ), 256, 0, stream>>>(out, gamma, beta);
}
